// model_52699248722070
// MI455X (gfx1250) — matrix operand order repaired and checked by reading, not compiled, not run on hardware
//
#include <hip/hip_runtime.h>
#include <math.h>

// ---------------- problem constants ----------------
#define V_NODES 50000
#define E_EDGES 25000
#define NNZ     600000
#define NS      8      // subspaces
#define NK      16     // dim per subspace
#define ND      128    // NS*NK
#define NUM_STEP 4
#define EPSF    1e-10f

typedef __attribute__((ext_vector_type(16))) _Float16 v16h;
typedef __attribute__((ext_vector_type(8)))  float    v8f;

// ---------------- workspace layout (in floats) ----------------
static constexpr size_t XOFF   = 0;
static constexpr size_t XCNT   = (size_t)V_NODES * ND;            // 6,400,000
static constexpr size_t YOFF   = XCNT + V_NODES;                  // 6,450,000
static constexpr size_t YCNT   = YOFF + (size_t)E_EDGES * ND;     // 9,650,000
static constexpr size_t GOFF   = YCNT + E_EDGES;                  // 9,675,000
static constexpr size_t CSOFF  = GOFF + NS * 256;                 // +2048
static constexpr size_t ENTOFF = CSOFF + ND;                      // +128
static constexpr size_t LOSSOFF= ENTOFF + 1;
// total ~9,677,179 floats (~38.7 MB)

// ---------------- kernels ----------------

__global__ void k_zero(float* __restrict__ p, long n) {
    long i = (long)blockIdx.x * blockDim.x + threadIdx.x;
    long stride = (long)gridDim.x * blockDim.x;
    for (; i < n; i += stride) p[i] = 0.0f;
}

// gumbel-softmax over K=16; one thread per (v,s) row
__global__ void k_gsoftmax(const float* __restrict__ emb,
                           const float* __restrict__ gum,
                           float* __restrict__ X) {
    int r = blockIdx.x * blockDim.x + threadIdx.x;
    if (r >= V_NODES * NS) return;
    size_t base = (size_t)r * NK;
    float v[NK];
    float m = -1e30f;
    #pragma unroll
    for (int k = 0; k < NK; ++k) {
        v[k] = emb[base + k] + gum[base + k];   // TAU == 1
        m = fmaxf(m, v[k]);
    }
    float sum = 0.0f;
    #pragma unroll
    for (int k = 0; k < NK; ++k) { v[k] = __expf(v[k] - m); sum += v[k]; }
    float inv = 1.0f / sum;
    #pragma unroll
    for (int k = 0; k < NK; ++k) X[base + k] = v[k] * inv;
}

// scatter-add of 128-float rows: one wave per nnz item, each lane does float4
__global__ __launch_bounds__(256) void k_scatter_add(
        const float* __restrict__ src, const int* __restrict__ gidx,
        const int* __restrict__ sidx, float* __restrict__ dst,
        float* __restrict__ cnt, int nnz) {
    int item = blockIdx.x * 8 + (threadIdx.x >> 5);
    int lane = threadIdx.x & 31;
    if (item >= nnz) return;
    int g = gidx[item];
    int d = sidx[item];
    float4 v = reinterpret_cast<const float4*>(src + (size_t)g * ND)[lane];
    float* db = dst + (size_t)d * ND + lane * 4;
    atomicAdd(db + 0, v.x);
    atomicAdd(db + 1, v.y);
    atomicAdd(db + 2, v.z);
    atomicAdd(db + 3, v.w);
    if (lane == 0) atomicAdd(cnt + d, 1.0f);
}

// divide each row by max(count,1): scatter_mean semantics
__global__ void k_normalize(float* __restrict__ dst, const float* __restrict__ cnt,
                            long nelem) {
    long i = (long)blockIdx.x * blockDim.x + threadIdx.x;
    if (i >= nelem) return;
    float c = cnt[i >> 7];
    dst[i] = dst[i] / fmaxf(c, 1.0f);
}

// fused entropy sum + column sums, block-reduced in LDS (ds_add_f32)
__global__ __launch_bounds__(256) void k_entropy_colsum(
        const float* __restrict__ Xsrc, int nrows,
        float* __restrict__ colsum, float* __restrict__ entAcc) {
    __shared__ float s_col[ND];
    __shared__ float s_ent;
    if (threadIdx.x < ND) s_col[threadIdx.x] = 0.0f;
    if (threadIdx.x == 0) s_ent = 0.0f;
    __syncthreads();
    int r = blockIdx.x * blockDim.x + threadIdx.x;
    if (r < nrows) {
        const float* row = Xsrc + (size_t)r * ND;
        float ent = 0.0f;
        #pragma unroll 8
        for (int d = 0; d < ND; ++d) {
            float v = row[d];
            atomicAdd(&s_col[d], v);
            ent -= v * __logf(v + EPSF);
        }
        atomicAdd(&s_ent, ent);
    }
    __syncthreads();
    if (threadIdx.x < ND) atomicAdd(&colsum[threadIdx.x], s_col[threadIdx.x]);
    if (threadIdx.x == 0) atomicAdd(entAcc, s_ent);
}

// ---------- WMMA Gram helpers ----------
// Full 32-row tile, branch-free. abase already includes the hi*8 row offset,
// bbase the hi*8 row offset (CDNA5 16-bit A/B fragment layouts), so the
// fragment build is pure load + cvt with no lane-dependent selects.
__device__ __forceinline__ v8f gram_tile(const float* __restrict__ abase,
                                         const float* __restrict__ bbase,
                                         int n0, v8f c) {
    float fa[16], fb[16];
    #pragma unroll
    for (int t = 0; t < 8; ++t) {
        fa[t]     = abase[(size_t)(n0 + t) * ND];
        fa[t + 8] = abase[(size_t)(n0 + 16 + t) * ND];
    }
    #pragma unroll
    for (int t = 0; t < 8; ++t) {
        fb[t]     = bbase[(size_t)(n0 + t) * ND];
        fb[t + 8] = bbase[(size_t)(n0 + 16 + t) * ND];
    }
    v16h a, b;
    #pragma unroll
    for (int t = 0; t < 16; ++t) {
        a[t] = (_Float16)fa[t];
        b[t] = (_Float16)fb[t];
    }
    return __builtin_amdgcn_wmma_f32_16x16x32_f16(
            false, a, false, b, (short)0, c, false, false);
}

// branchless clamped load for the tail tile: row relative to colptr, rem valid rows
__device__ __forceinline__ float ld_clamp(const float* __restrict__ colptr,
                                          int row, int rem) {
    int rc = row < rem ? row : (rem - 1);
    float v = colptr[(size_t)rc * ND];
    return row < rem ? v : 0.0f;
}

__device__ __forceinline__ v8f gram_tile_tail(const float* __restrict__ colptr,
                                              int hi, int rem, v8f c) {
    float fa[16], fb[16];
    #pragma unroll
    for (int t = 0; t < 8; ++t) {
        fa[t]     = ld_clamp(colptr, hi * 8 + t, rem);
        fa[t + 8] = ld_clamp(colptr, hi * 8 + 16 + t, rem);
    }
    #pragma unroll
    for (int t = 0; t < 8; ++t) {
        fb[t]     = ld_clamp(colptr, hi * 8 + t, rem);
        fb[t + 8] = ld_clamp(colptr, hi * 8 + 16 + t, rem);
    }
    v16h a, b;
    #pragma unroll
    for (int t = 0; t < 16; ++t) {
        a[t] = (_Float16)fa[t];
        b[t] = (_Float16)fb[t];
    }
    return __builtin_amdgcn_wmma_f32_16x16x32_f16(
            false, a, false, b, (short)0, c, false, false);
}

// Gram matrices G_s = X_s^T X_s via V_WMMA_F32_16X16X32_F16.
// grid = NS * NCHUNK single-wave blocks; dual accumulators, 64 rows/iter.
#define NCHUNK 32
__global__ __launch_bounds__(32) void k_gram_wmma(
        const float* __restrict__ Xsrc, int nrows, float* __restrict__ G) {
    int s     = blockIdx.x & (NS - 1);
    int chunk = blockIdx.x >> 3;            // / NS
    int rpc   = (((nrows + NCHUNK - 1) / NCHUNK) + 31) & ~31;  // multiple of 32
    int r0 = chunk * rpc;
    int r1 = r0 + rpc; if (r1 > nrows) r1 = nrows;
    int lane = threadIdx.x;
    int j  = lane & 15;
    int hi = lane >> 4;
    const float* col0  = Xsrc + (size_t)s * NK + j;
    const float* abase = col0 + (size_t)(hi * 8)  * ND;  // A rows: K = hi*8  + {0..7,16..23}
    const float* bbase = col0 + (size_t)(hi * 8) * ND;   // B rows: K = hi*8  + {0..7,16..23}
    v8f c0 = {}, c1 = {};
    int n0 = r0;
    for (; n0 + 64 <= r1; n0 += 64) {
        c0 = gram_tile(abase, bbase, n0, c0);
        c1 = gram_tile(abase, bbase, n0 + 32, c1);
    }
    if (n0 + 32 <= r1) {
        c0 = gram_tile(abase, bbase, n0, c0);
        n0 += 32;
    }
    if (n0 < r1) {
        c1 = gram_tile_tail(col0 + (size_t)n0 * ND, hi, r1 - n0, c1);
    }
    v8f c = c0 + c1;
    // C layout: VGPR r -> M = r + 8*hi, N = j
    #pragma unroll
    for (int r = 0; r < 8; ++r) {
        int M = r + hi * 8;
        atomicAdd(&G[s * 256 + M * 16 + j], c[r]);
    }
}

// one block, 128 threads: softmax rows of C = G/outer(norm,norm), losses
__global__ __launch_bounds__(128) void k_finalize(
        const float* __restrict__ G, const float* __restrict__ colsum,
        const float* __restrict__ entAcc, float nrows_f,
        float* __restrict__ lossAcc) {
    __shared__ float s_glob;
    if (threadIdx.x == 0) s_glob = 0.0f;
    __syncthreads();
    int t = threadIdx.x;            // (s, i) row of a 16x16 C matrix
    int s = t >> 4, i = t & 15;
    const float* Gs = G + s * 256;
    float normi = sqrtf(fmaxf(Gs[i * 16 + i], 0.0f));
    float Crow[16];
    float m = -1e30f;
    #pragma unroll
    for (int jj = 0; jj < 16; ++jj) {
        float normj = sqrtf(fmaxf(Gs[jj * 16 + jj], 0.0f));
        float cv = Gs[i * 16 + jj] / fmaxf(normi * normj, EPSF);
        Crow[jj] = cv;
        m = fmaxf(m, cv);
    }
    float sum = 0.0f;
    #pragma unroll
    for (int jj = 0; jj < 16; ++jj) sum += __expf(Crow[jj] - m);
    float logPii = (Crow[i] - m) - __logf(sum);
    float disc = -logPii * (1.0f / 128.0f);              // mean over (S,K)
    float p = colsum[t] / nrows_f;
    float gterm = p * __logf(p + EPSF) * (1.0f / (float)NS); // == -mean_s entropy_s
    atomicAdd(&s_glob, disc + gterm);
    __syncthreads();
    if (threadIdx.x == 0) {
        atomicAdd(&lossAcc[0], entAcc[0] / (nrows_f * (float)NS));
        atomicAdd(&lossAcc[1], s_glob);
    }
}

__global__ void k_writeout(const float* __restrict__ lossAcc, float* __restrict__ out) {
    if (threadIdx.x == 0) { out[0] = lossAcc[0]; out[1] = lossAcc[1]; }
}

// ---------------- host launcher ----------------
extern "C" void kernel_launch(void* const* d_in, const int* in_sizes, int n_in,
                              void* d_out, int out_size, void* d_ws, size_t ws_size,
                              hipStream_t stream) {
    (void)in_sizes; (void)n_in; (void)out_size; (void)ws_size;
    const float* emb  = (const float*)d_in[0];
    const float* gum  = (const float*)d_in[1];
    const int*   Vidx = (const int*)d_in[2];
    const int*   Eidx = (const int*)d_in[3];
    float* ws   = (float*)d_ws;
    float* X    = ws + XOFF;
    float* cntV = ws + XCNT;
    float* Y    = ws + YOFF;
    float* cntE = ws + YCNT;
    float* G    = ws + GOFF;
    float* cs   = ws + CSOFF;
    float* ent  = ws + ENTOFF;
    float* loss = ws + LOSSOFF;
    float* out  = (float*)d_out;

    auto zgrid = [](long n) { return (unsigned)((n + 255) / 256); };

    k_zero<<<1, 64, 0, stream>>>(loss, 2);
    k_gsoftmax<<<(V_NODES * NS + 255) / 256, 256, 0, stream>>>(emb, gum, X);

    for (int step = 0; step < NUM_STEP; ++step) {
        // ---- Y = scatter_mean(X[V_idx], E_idx) ----
        k_zero<<<zgrid((long)E_EDGES * ND + E_EDGES), 256, 0, stream>>>(
            Y, (long)E_EDGES * ND + E_EDGES);                 // Y + cntE contiguous
        k_scatter_add<<<(NNZ + 7) / 8, 256, 0, stream>>>(X, Vidx, Eidx, Y, cntE, NNZ);
        k_normalize<<<zgrid((long)E_EDGES * ND), 256, 0, stream>>>(
            Y, cntE, (long)E_EDGES * ND);
        k_zero<<<16, 256, 0, stream>>>(G, NS * 256 + ND + 1); // G + colsum + ent
        k_entropy_colsum<<<(E_EDGES + 255) / 256, 256, 0, stream>>>(Y, E_EDGES, cs, ent);
        k_gram_wmma<<<NS * NCHUNK, 32, 0, stream>>>(Y, E_EDGES, G);
        k_finalize<<<1, 128, 0, stream>>>(G, cs, ent, (float)E_EDGES, loss);

        // ---- X = scatter_mean(Y[E_idx], V_idx) ----
        k_zero<<<zgrid((long)V_NODES * ND + V_NODES), 256, 0, stream>>>(
            X, (long)V_NODES * ND + V_NODES);                 // X + cntV contiguous
        k_scatter_add<<<(NNZ + 7) / 8, 256, 0, stream>>>(Y, Eidx, Vidx, X, cntV, NNZ);
        k_normalize<<<zgrid((long)V_NODES * ND), 256, 0, stream>>>(
            X, cntV, (long)V_NODES * ND);
        k_zero<<<16, 256, 0, stream>>>(G, NS * 256 + ND + 1);
        k_entropy_colsum<<<(V_NODES + 255) / 256, 256, 0, stream>>>(X, V_NODES, cs, ent);
        k_gram_wmma<<<NS * NCHUNK, 32, 0, stream>>>(X, V_NODES, G);
        k_finalize<<<1, 128, 0, stream>>>(G, cs, ent, (float)V_NODES, loss);
    }
    k_writeout<<<1, 32, 0, stream>>>(loss, out);
}
